// NLayerGat_2035814498362
// MI455X (gfx1250) — matrix operand order repaired and checked by reading, not compiled, not run on hardware
//
#include <hip/hip_runtime.h>
#include <math.h>
#include <stdint.h>

#define NNODES 50000
#define NEDGES 800000
#define ETOT   (NEDGES + NNODES)
#define FEAT   128
#define NHEADS 8
#define FHD    16
#define NCLS   16

typedef __attribute__((ext_vector_type(16))) __bf16 v16bf;
typedef __attribute__((ext_vector_type(8)))  __bf16 v8bf;
typedef __attribute__((ext_vector_type(8)))  float  v8f;

// ---------- helpers ----------

__device__ __forceinline__ unsigned f2ord(float f) {
    unsigned u = __float_as_uint(f);
    return (u & 0x80000000u) ? ~u : (u | 0x80000000u);
}
__device__ __forceinline__ float ord2f(unsigned u) {
    return (u & 0x80000000u) ? __uint_as_float(u & 0x7fffffffu) : __uint_as_float(~u);
}
// f2ord(-inf) == 0x007FFFFF
#define NEG_INF_ORD 0x007FFFFFu

__device__ __forceinline__ void edge_ends(const int* __restrict__ src,
                                          const int* __restrict__ dst,
                                          long long e, int& s, int& d) {
    if (e < NEDGES) { s = src[e]; d = dst[e]; }
    else            { s = d = (int)(e - NEDGES); }   // implicit self loops
}

// CDNA5 async global->LDS copy of 16 bytes (ASYNCcnt-tracked).
__device__ __forceinline__ void async_copy_b128(const __bf16* gsrc, __bf16* ldst) {
    unsigned laddr = (unsigned)(uintptr_t)ldst;                 // low 32 bits = LDS offset
    unsigned long long gaddr = (unsigned long long)(uintptr_t)gsrc;
    asm volatile("global_load_async_to_lds_b128 %0, %1, off"
                 :: "v"(laddr), "v"(gaddr) : "memory");
}
__device__ __forceinline__ void async_wait0() {
    asm volatile("s_wait_asynccnt 0x0" ::: "memory");
}

// ---------- kernels ----------

// h_bf16[n,:] = (bf16) emb[x[n], :]
__global__ void gather_embed_kernel(const float* __restrict__ emb,
                                    const int* __restrict__ x,
                                    __bf16* __restrict__ h) {
    int idx = blockIdx.x * blockDim.x + threadIdx.x;      // one float4 each
    int tot = NNODES * (FEAT / 4);
    if (idx >= tot) return;
    int node = idx >> 5;            // FEAT/4 == 32
    int c4   = idx & 31;
    float4 v = reinterpret_cast<const float4*>(emb + (size_t)x[node] * FEAT)[c4];
    __bf16* o = h + (size_t)node * FEAT + c4 * 4;
    o[0] = (__bf16)v.x; o[1] = (__bf16)v.y; o[2] = (__bf16)v.z; o[3] = (__bf16)v.w;
}

// WbfT[c, k] = (bf16) W[k, c]   (W is [128, ncols] row-major fp32)
__global__ void convert_w_kernel(const float* __restrict__ W,
                                 __bf16* __restrict__ WbfT, int ncols) {
    int idx = blockIdx.x * blockDim.x + threadIdx.x;
    int tot = FEAT * ncols;
    if (idx >= tot) return;
    int k = idx / ncols, c = idx % ncols;
    WbfT[(size_t)c * FEAT + k] = (__bf16)W[idx];
}

// C[N, ncols] = A_bf16[N, 128] @ W (WbfT is [ncols, 128] bf16).
// Block = 8 waves. ncols==128: block does 16 rows x 128 cols (wave = col tile).
// ncols==16: block does 128 rows x 16 cols (wave = row tile).
// A tile and W are staged through LDS with CDNA5 async loads; compute is
// ds_load_b128 fragments + 4x v_wmma_f32_16x16x32_bf16.
__global__ __launch_bounds__(256) void gemm_bf16_wmma_kernel(
        const __bf16* __restrict__ Abf,
        const __bf16* __restrict__ WbfT,
        float* __restrict__ C, int ncols) {
    __shared__ __bf16 smem[18432];                 // 36 KB: W tile + A tile
    const int tid  = threadIdx.x;
    const int lane = tid & 31;
    const int wave = tid >> 5;

    const int wsElems = ncols * FEAT;              // 16384 (32KB) or 2048 (4KB)
    __bf16* Ws = smem;
    __bf16* As = smem + wsElems;

    int aRows, rowBase, tileLocalRow, nTile;
    if (ncols == FEAT) { aRows = 16;  rowBase = blockIdx.x * 16;  tileLocalRow = 0;         nTile = wave; }
    else               { aRows = 128; rowBase = blockIdx.x * 128; tileLocalRow = wave * 16; nTile = 0;    }

    // Stage W (shared by all waves in block)
    {
        int chunks = wsElems / 8;                  // 16B chunks
        for (int c = tid; c < chunks; c += 256)
            async_copy_b128(WbfT + c * 8, Ws + c * 8);
    }
    // Stage A tile (clamp OOB rows; clamped rows are never consumed)
    {
        int chunks = aRows * (FEAT / 8);
        for (int c = tid; c < chunks; c += 256) {
            int r = c >> 4, cc = c & 15;
            int grow = rowBase + r;
            if (grow >= NNODES) grow = 0;
            async_copy_b128(Abf + (size_t)grow * FEAT + cc * 8, As + c * 8);
        }
    }
    async_wait0();
    __syncthreads();

    int tileRowGlobal = rowBase + tileLocalRow;
    if (tileRowGlobal < NNODES) {                  // wave-uniform; EXEC all-1 inside
        int m  = lane & 15;                        // A row in tile / B column in tile
        int hh = lane >> 4;
        const __bf16* Ap = As + (size_t)(tileLocalRow + m) * FEAT;
        const __bf16* Bp = Ws + (size_t)(nTile * 16 + m) * FEAT;

        v8f acc = {};
        #pragma unroll
        for (int k0 = 0; k0 < FEAT; k0 += 32) {
            // A 16x32 bf16 fragment: lanes 0-15 K=[8h,8h+8)+[16+8h,..), element order
            // is K-sequential within each 8-element half (ISA 7.12.2).
            v8bf alo = *reinterpret_cast<const v8bf*>(Ap + k0 + 8 * hh);
            v8bf ahi = *reinterpret_cast<const v8bf*>(Ap + k0 + 16 + 8 * hh);
            v16bf a = __builtin_shufflevector(alo, ahi,
                0, 1, 2, 3, 4, 5, 6, 7, 8, 9, 10, 11, 12, 13, 14, 15);
            // B 32x16 fragment: same K order as A: K=[8h,8h+8)+[16+8h,16+8h+8).
            v8bf blo = *reinterpret_cast<const v8bf*>(Bp + k0 + 8 * hh);
            v8bf bhi = *reinterpret_cast<const v8bf*>(Bp + k0 + 16 + 8 * hh);
            v16bf b = __builtin_shufflevector(blo, bhi,
                0, 1, 2, 3, 4, 5, 6, 7, 8, 9, 10, 11, 12, 13, 14, 15);
            acc = __builtin_amdgcn_wmma_f32_16x16x32_bf16(
                false, a, false, b, (short)0, acc, false, false);
        }
        #pragma unroll
        for (int r = 0; r < 8; ++r)
            C[(size_t)(tileRowGlobal + r + 8 * hh) * ncols + nTile * 16 + m] = acc[r];
    }
}

__global__ void att_logits_kernel(const float* __restrict__ hW,
                                  const float* __restrict__ a_src,
                                  const float* __restrict__ a_dst,
                                  float* __restrict__ als,
                                  float* __restrict__ ald,
                                  int heads) {
    int idx = blockIdx.x * blockDim.x + threadIdx.x;
    int tot = NNODES * heads;
    if (idx >= tot) return;
    int node = idx / heads;
    int h    = idx % heads;
    const float* p  = hW + (size_t)node * heads * FHD + h * FHD;
    const float* as = a_src + h * FHD;
    const float* ad = a_dst + h * FHD;
    float ss = 0.f, dd = 0.f;
    #pragma unroll
    for (int f = 0; f < FHD; ++f) { float v = p[f]; ss += v * as[f]; dd += v * ad[f]; }
    als[idx] = ss;
    ald[idx] = dd;
}

__global__ void fill_u32_kernel(unsigned* __restrict__ p, unsigned v, long long count) {
    long long idx = blockIdx.x * (long long)blockDim.x + threadIdx.x;
    if (idx < count) p[idx] = v;
}

__global__ void edge_max_kernel(const int* __restrict__ src, const int* __restrict__ dst,
                                const float* __restrict__ als, const float* __restrict__ ald,
                                unsigned* __restrict__ m_u, int heads) {
    long long idx = blockIdx.x * (long long)blockDim.x + threadIdx.x;
    long long tot = (long long)ETOT * heads;
    if (idx >= tot) return;
    long long e = idx / heads;
    int h = (int)(idx % heads);
    int s, d; edge_ends(src, dst, e, s, d);
    float v = als[(size_t)s * heads + h] + ald[(size_t)d * heads + h];
    v = (v > 0.f) ? v : 0.2f * v;                       // leaky_relu(0.2)
    atomicMax(&m_u[(size_t)d * heads + h], f2ord(v));
}

__global__ void edge_expsum_kernel(const int* __restrict__ src, const int* __restrict__ dst,
                                   const float* __restrict__ als, const float* __restrict__ ald,
                                   const unsigned* __restrict__ m_u,
                                   float* __restrict__ e_edge, float* __restrict__ ssum,
                                   int heads) {
    long long idx = blockIdx.x * (long long)blockDim.x + threadIdx.x;
    long long tot = (long long)ETOT * heads;
    if (idx >= tot) return;
    long long e = idx / heads;
    int h = (int)(idx % heads);
    int s, d; edge_ends(src, dst, e, s, d);
    float v = als[(size_t)s * heads + h] + ald[(size_t)d * heads + h];
    v = (v > 0.f) ? v : 0.2f * v;
    float ex = __expf(v - ord2f(m_u[(size_t)d * heads + h]));
    e_edge[idx] = ex;
    atomicAdd(&ssum[(size_t)d * heads + h], ex);
}

__global__ void edge_agg_kernel(const int* __restrict__ src, const int* __restrict__ dst,
                                const float* __restrict__ e_edge,
                                const float* __restrict__ ssum,
                                const float* __restrict__ hW,
                                float* __restrict__ agg, int heads) {
    long long idx = blockIdx.x * (long long)blockDim.x + threadIdx.x;
    long long tot = (long long)ETOT * heads;
    if (idx >= tot) return;
    long long e = idx / heads;
    int h = (int)(idx % heads);
    int s, d; edge_ends(src, dst, e, s, d);
    float alpha = e_edge[idx] / (ssum[(size_t)d * heads + h] + 1e-16f);
    const float* hp = hW + (size_t)s * heads * FHD + h * FHD;
    float* ap = agg + (size_t)d * heads * FHD + h * FHD;
    #pragma unroll
    for (int f = 0; f < FHD; ++f) atomicAdd(&ap[f], hp[f] * alpha);
}

// out_bf16 = (bf16) leaky_relu(agg + b, 0.01)  -- feeds next layer's GEMM only
__global__ void bias_act_kernel(const float* __restrict__ agg, const float* __restrict__ b,
                                __bf16* __restrict__ out, int cols) {
    int idx = blockIdx.x * blockDim.x + threadIdx.x;
    int tot = NNODES * cols;
    if (idx >= tot) return;
    int c = idx % cols;
    float v = agg[idx] + b[c];
    out[idx] = (__bf16)((v > 0.f) ? v : 0.01f * v);     // F.leaky_relu default
}

__global__ void logsoftmax_kernel(const float* __restrict__ agg, const float* __restrict__ b,
                                  float* __restrict__ out) {
    int node = blockIdx.x * blockDim.x + threadIdx.x;
    if (node >= NNODES) return;
    float v[NCLS];
    float mx = -INFINITY;
    #pragma unroll
    for (int c = 0; c < NCLS; ++c) {
        v[c] = agg[(size_t)node * NCLS + c] + b[c];
        mx = fmaxf(mx, v[c]);
    }
    float s = 0.f;
    #pragma unroll
    for (int c = 0; c < NCLS; ++c) s += __expf(v[c] - mx);
    float ls = __logf(s);
    #pragma unroll
    for (int c = 0; c < NCLS; ++c) out[(size_t)node * NCLS + c] = v[c] - mx - ls;
}

// ---------- host orchestration ----------

static inline int cdiv(long long a, long long b) { return (int)((a + b - 1) / b); }

static void run_gat_layer(const __bf16* hin_bf, const float* W,
                          const float* a_src, const float* a_dst,
                          int heads,
                          const int* src, const int* dst,
                          __bf16* WbfT, float* hW, float* als, float* ald,
                          unsigned* m_u, float* ssum, float* e_edge, float* agg,
                          hipStream_t stream) {
    int ncols = heads * FHD;

    convert_w_kernel<<<cdiv((long long)FEAT * ncols, 256), 256, 0, stream>>>(W, WbfT, ncols);

    int gemmBlocks = (ncols == FEAT) ? (NNODES / 16) : cdiv(NNODES, 128);
    gemm_bf16_wmma_kernel<<<gemmBlocks, 256, 0, stream>>>(hin_bf, WbfT, hW, ncols);

    att_logits_kernel<<<cdiv((long long)NNODES * heads, 256), 256, 0, stream>>>(
        hW, a_src, a_dst, als, ald, heads);

    long long nh = (long long)NNODES * heads;
    fill_u32_kernel<<<cdiv(nh, 256), 256, 0, stream>>>(m_u, NEG_INF_ORD, nh);
    fill_u32_kernel<<<cdiv(nh, 256), 256, 0, stream>>>((unsigned*)ssum, 0u, nh);
    long long nagg = (long long)NNODES * ncols;
    fill_u32_kernel<<<cdiv(nagg, 256), 256, 0, stream>>>((unsigned*)agg, 0u, nagg);

    long long eh = (long long)ETOT * heads;
    edge_max_kernel<<<cdiv(eh, 256), 256, 0, stream>>>(src, dst, als, ald, m_u, heads);
    edge_expsum_kernel<<<cdiv(eh, 256), 256, 0, stream>>>(src, dst, als, ald, m_u, e_edge, ssum, heads);
    edge_agg_kernel<<<cdiv(eh, 256), 256, 0, stream>>>(src, dst, e_edge, ssum, hW, agg, heads);
}

extern "C" void kernel_launch(void* const* d_in, const int* in_sizes, int n_in,
                              void* d_out, int out_size, void* d_ws, size_t ws_size,
                              hipStream_t stream) {
    (void)in_sizes; (void)n_in; (void)out_size; (void)ws_size;

    const int*   x   = (const int*)d_in[0];
    const int*   ei  = (const int*)d_in[1];
    const int*   src = ei;
    const int*   dst = ei + NEDGES;
    const float* emb = (const float*)d_in[2];
    const float* W0  = (const float*)d_in[3];
    const float* as0 = (const float*)d_in[4];
    const float* ad0 = (const float*)d_in[5];
    const float* b0  = (const float*)d_in[6];
    const float* W1  = (const float*)d_in[7];
    const float* as1 = (const float*)d_in[8];
    const float* ad1 = (const float*)d_in[9];
    const float* b1  = (const float*)d_in[10];
    const float* W2  = (const float*)d_in[11];
    const float* as2 = (const float*)d_in[12];
    const float* ad2 = (const float*)d_in[13];
    const float* b2  = (const float*)d_in[14];
    float* out = (float*)d_out;

    float* ws = (float*)d_ws;
    size_t o = 0;
    float*    hW     = ws + o; o += (size_t)NNODES * FEAT;
    float*    agg    = ws + o; o += (size_t)NNODES * FEAT;
    __bf16*   hbuf   = (__bf16*)(ws + o); o += (size_t)NNODES * FEAT / 2;   // bf16 [N,128]
    __bf16*   WbfT   = (__bf16*)(ws + o); o += (size_t)FEAT * FEAT / 2;     // bf16 [ncols,128]
    float*    als    = ws + o; o += (size_t)NNODES * NHEADS;
    float*    ald    = ws + o; o += (size_t)NNODES * NHEADS;
    unsigned* m_u    = (unsigned*)(ws + o); o += (size_t)NNODES * NHEADS;
    float*    ssum   = ws + o; o += (size_t)NNODES * NHEADS;
    float*    e_edge = ws + o; o += (size_t)ETOT * NHEADS;

    // h = (bf16) emb[x]
    gather_embed_kernel<<<cdiv((long long)NNODES * (FEAT / 4), 256), 256, 0, stream>>>(emb, x, hbuf);

    // Layer 0: GAT(128 -> 8x16, concat) + leaky_relu(0.01)
    run_gat_layer(hbuf, W0, as0, ad0, NHEADS, src, dst, WbfT, hW, als, ald, m_u, ssum, e_edge, agg, stream);
    bias_act_kernel<<<cdiv((long long)NNODES * FEAT, 256), 256, 0, stream>>>(agg, b0, hbuf, FEAT);

    // Layer 1
    run_gat_layer(hbuf, W1, as1, ad1, NHEADS, src, dst, WbfT, hW, als, ald, m_u, ssum, e_edge, agg, stream);
    bias_act_kernel<<<cdiv((long long)NNODES * FEAT, 256), 256, 0, stream>>>(agg, b1, hbuf, FEAT);

    // Layer 2: single head, 16 classes, mean over 1 head == identity
    run_gat_layer(hbuf, W2, as2, ad2, 1, src, dst, WbfT, hW, als, ald, m_u, ssum, e_edge, agg, stream);

    // log_softmax(agg + b2)
    logsoftmax_kernel<<<cdiv(NNODES, 256), 256, 0, stream>>>(agg, b2, out);
}
